// INT4PackedLinear_64330020159897
// MI455X (gfx1250) — matrix operand order repaired and checked by reading, not compiled, not run on hardware
//
#include <hip/hip_runtime.h>

typedef _Float16 v16h __attribute__((ext_vector_type(16)));
typedef _Float16 v8h  __attribute__((ext_vector_type(8)));
typedef _Float16 h2   __attribute__((ext_vector_type(2)));
typedef float    v8f  __attribute__((ext_vector_type(8)));
typedef unsigned int u32;

union AU { v16h v; v8h h[2]; };
union BU { v16h v; h2 p[8]; };

// Problem shape (fixed by the reference setup)
#define MM 64
#define KK 8192
#define NN 28672
#define GG 128
#define NGROUPS (KK / GG)   // 64
#define LDS_PITCH 136       // 128 + 8 halves padding -> conflict-free banks

__global__ __launch_bounds__(256, 1)
void int4_packed_linear_kernel(const _Float16* __restrict__ x,
                               const int* __restrict__ wp,
                               const _Float16* __restrict__ scales,
                               const _Float16* __restrict__ bias,
                               _Float16* __restrict__ y) {
  __shared__ __align__(16) _Float16 xs[2][MM][LDS_PITCH];

  const int tid  = threadIdx.x;
  const int lane = tid & 31;
  const int wave = tid >> 5;
  const int nl   = lane & 15;      // M row within tile / column within strip
  const int hsel = lane >> 4;      // 0: K 0-15 half, 1: K 16-31 half
  const int col  = (blockIdx.x * 8 + wave) * 16 + nl;

  // Per-lane weight base: column col; lanes 16-31 start 8 packed rows down
  const int* wb = wp + (size_t)hsel * 8 * NN + col;

  v8f acc[4] = {};
  const h2 c1032 = {(_Float16)-1032.0f, (_Float16)-1032.0f};

  // ---- helpers -------------------------------------------------------------
  // Issue global loads of x for k-group g into staging regs (4 x 16B/thread).
  auto issue_stage = [&](int g, uint4* sreg) {
    #pragma unroll
    for (int q = 0; q < 4; ++q) {
      const int c = tid + 256 * q;           // 1024 chunks of 8 halves
      const int m = c >> 4, ko = (c & 15) * 8;
      sreg[q] = *(const uint4*)(x + (size_t)m * KK + (size_t)g * GG + ko);
    }
  };
  // Write staged x into LDS buffer bufi (ds_store_b128, coalesced rows).
  auto store_stage = [&](int bufi, const uint4* sreg) {
    #pragma unroll
    for (int q = 0; q < 4; ++q) {
      const int c = tid + 256 * q;
      const int m = c >> 4, ko = (c & 15) * 8;
      *(uint4*)&xs[bufi][m][ko] = sreg[q];
    }
  };
  // Issue all 32 NT weight dword loads for k-group g (4 steps x 8 rows).
  auto issue_w = [&](int g, int* w) {
    const int* p = wb + (size_t)g * (GG / 2) * NN;
    #pragma unroll
    for (int i = 0; i < 4; ++i)
      #pragma unroll
      for (int j = 0; j < 8; ++j)
        w[i * 8 + j] = __builtin_nontemporal_load(p + (size_t)(i * 16 + j) * NN);
  };
  // Dequant + 16 WMMAs for one k-group out of LDS buffer bufi.
  auto compute_group = [&](int bufi, const int* w, _Float16 s) {
    const h2 s2 = {s, s};
    #pragma unroll
    for (int i = 0; i < 4; ++i) {
      BU b;
      #pragma unroll
      for (int j = 0; j < 8; ++j) {
        u32 v = (u32)w[i * 8 + j];
        u32 m = ((v | (v << 12)) & 0x000F000Fu) | 0x64006400u; // {1024+lo,1024+hi}
        h2 hh = __builtin_bit_cast(h2, m);
        hh = hh + c1032;      // exact: values in [-8, 7]
        hh = hh * s2;         // single f16 rounding = reference dequant
        b.p[j] = hh;
      }
      const int kl = i * 32 + hsel * 16;
      #pragma unroll
      for (int t = 0; t < 4; ++t) {
        AU a;
        a.h[0] = *(const v8h*)&xs[bufi][t * 16 + nl][kl];       // ds_load_b128
        a.h[1] = *(const v8h*)&xs[bufi][t * 16 + nl][kl + 8];  // ds_load_b128
        acc[t] = __builtin_amdgcn_wmma_f32_16x16x32_f16(
            false, a.v, false, b.v, (short)0, acc[t], false, false);
      }
    }
  };
  // --------------------------------------------------------------------------

  uint4 sreg[4];
  int w0[32], w1[32];

  // Prologue: stage group 0, prefetch weights+scale for group 0, stage group 1.
  issue_stage(0, sreg);
  store_stage(0, sreg);
  issue_w(0, w0);
  _Float16 s = scales[col];
  __syncthreads();
  issue_stage(1, sreg);

  for (int g = 0; g < NGROUPS; g += 2) {
    // ---- even group g: weights in w0, x in buf 0 ----
    {
      issue_w(g + 1, w1);                                   // g+1 <= 63 always
      _Float16 sn = scales[(size_t)(g + 1) * NN + col];
      compute_group(0, w0, s);
      store_stage(1, sreg);                                 // x of group g+1
      __syncthreads();
      if (g + 2 < NGROUPS) issue_stage(g + 2, sreg);
      s = sn;
    }
    // ---- odd group g+1: weights in w1, x in buf 1 ----
    {
      const int gn = (g + 2 < NGROUPS) ? g + 2 : g + 1;     // clamp at the end
      issue_w(gn, w0);
      _Float16 sn = scales[(size_t)gn * NN + col];
      compute_group(1, w1, s);
      if (g + 2 < NGROUPS) store_stage(0, sreg);            // x of group g+2
      __syncthreads();
      if (g + 3 < NGROUPS) issue_stage(g + 3, sreg);
      s = sn;
    }
  }

  // Epilogue: f16 convert, f16 bias add (matches reference), store.
  const _Float16 bcol = bias[col];
  #pragma unroll
  for (int t = 0; t < 4; ++t) {
    #pragma unroll
    for (int r = 0; r < 8; ++r) {
      const int m = t * 16 + hsel * 8 + r;  // C/D layout: vgpr r, lane half -> M
      y[(size_t)m * NN + col] = (_Float16)acc[t][r] + bcol;
    }
  }
}

extern "C" void kernel_launch(void* const* d_in, const int* in_sizes, int n_in,
                              void* d_out, int out_size, void* d_ws, size_t ws_size,
                              hipStream_t stream) {
  (void)in_sizes; (void)n_in; (void)out_size; (void)d_ws; (void)ws_size;
  const _Float16* x      = (const _Float16*)d_in[0];
  const int*      wp     = (const int*)d_in[1];
  const _Float16* scales = (const _Float16*)d_in[2];
  const _Float16* bias   = (const _Float16*)d_in[3];
  _Float16*       y      = (_Float16*)d_out;

  dim3 grid(NN / 128);   // 8 waves/block * 16 cols/wave = 128 cols per block
  dim3 block(256);
  hipLaunchKernelGGL(int4_packed_linear_kernel, grid, block, 0, stream,
                     x, wp, scales, bias, y);
}
